// Generative_30631706755300
// MI455X (gfx1250) — matrix operand order repaired and checked by reading, not compiled, not run on hardware
//
#include <hip/hip_runtime.h>
#include <hip/hip_bf16.h>
#include <math.h>

// Problem constants
#define B_   8192
#define IN_  784
#define H_   1024
#define L_   128
#define T_   10

// GEMM tiling: 64(M) x 128(N) x 32(K) macrotile, 8 waves of 32, each wave 32x32
#define TILE_M 64
#define TILE_N 128
#define TILE_K 32
#define LDT    40                        // LDS row stride in halves (80B, 16B aligned)
#define BPAD   (B_ + T_*TILE_M)          // 8832 padded rows
#define MAX_ROWTILES (B_/TILE_M + T_)    // 138 worst-case row tiles

// gfx1250 async global->LDS path (guarded: falls back to sync copies if absent)
#ifndef __has_builtin
#define __has_builtin(x) 0
#endif
#if __has_builtin(__builtin_amdgcn_global_load_async_to_lds_b128)
#define HAVE_ASYNC_LDS 1
#else
#define HAVE_ASYNC_LDS 0
#endif

typedef __attribute__((ext_vector_type(16))) _Float16 v16h;
typedef __attribute__((ext_vector_type(8)))  _Float16 v8h;
typedef __attribute__((ext_vector_type(8)))  float    v8f;
typedef __attribute__((ext_vector_type(4)))  float    v4f;

#if HAVE_ASYNC_LDS
// Builtin signature (from hipcc diagnostic): param type is
//   __attribute__((__vector_size__(4*sizeof(int)))) int __device__ *   (global int4*)
typedef int v4i_vs __attribute__((vector_size(4 * sizeof(int))));
__device__ __forceinline__ void async_cp16(const _Float16* g, _Float16* l) {
  __builtin_amdgcn_global_load_async_to_lds_b128(
      (__attribute__((address_space(1))) v4i_vs*)g,
      (__attribute__((address_space(3))) v4i_vs*)l, 0, 0);
}
#endif

__device__ __forceinline__ void wait_async0() {
#if HAVE_ASYNC_LDS
#if __has_builtin(__builtin_amdgcn_s_wait_asynccnt)
  __builtin_amdgcn_s_wait_asynccnt(0);
#else
  asm volatile("s_wait_asynccnt 0x0" ::: "memory");
#endif
#endif
}

// ---------------- setup kernels ----------------

__global__ void k_init(int* counts, int* cursor, int* perm) {
  int tid = threadIdx.x;
  if (tid < T_) { counts[tid] = 0; cursor[tid] = 0; }
  for (int i = tid; i < BPAD; i += blockDim.x) perm[i] = -1;
}

__global__ void k_count(const int* __restrict__ task, int* counts) {
  int b = blockIdx.x * blockDim.x + threadIdx.x;
  if (b < B_) atomicAdd(&counts[task[b]], 1);
}

__global__ void k_scan(const int* __restrict__ counts, int* offsets) {
  if (threadIdx.x == 0 && blockIdx.x == 0) {
    int off = 0;
    offsets[0] = 0;
    for (int t = 0; t < T_; ++t) {
      off += ((counts[t] + TILE_M - 1) / TILE_M) * TILE_M;
      offsets[t + 1] = off;
    }
  }
}

__global__ void k_scatter(const int* __restrict__ task, const int* __restrict__ offsets,
                          int* cursor, int* perm) {
  int b = blockIdx.x * blockDim.x + threadIdx.x;
  if (b < B_) {
    int t = task[b];
    int pos = atomicAdd(&cursor[t], 1);
    perm[offsets[t] + pos] = b;
  }
}

// ---------------- weight pre-transpose: f32 (K,N) -> f16 (N,K) ----------------
// grid: (ceil(N/32), ceil(K/32), tasks), block (32,8)

__global__ __launch_bounds__(256) void k_transpose(const float* __restrict__ W,
                                                   _Float16* __restrict__ WT,
                                                   int K, int N)
{
  __shared__ float tile[32][33];
  const long mat = (long)blockIdx.z * K * N;
  const float*   Wz  = W  + mat;
  _Float16*      WTz = WT + mat;
  int n0 = blockIdx.x * 32;
  int k0 = blockIdx.y * 32;
  int tx = threadIdx.x, ty = threadIdx.y;
  #pragma unroll
  for (int r = 0; r < 4; ++r) {
    int k = k0 + ty + r * 8;
    int n = n0 + tx;
    tile[ty + r * 8][tx] = (k < K && n < N) ? Wz[(long)k * N + n] : 0.f;
  }
  __syncthreads();
  #pragma unroll
  for (int r = 0; r < 4; ++r) {
    int n = n0 + ty + r * 8;
    int k = k0 + tx;
    if (n < N && k < K) WTz[(long)n * K + k] = (_Float16)tile[tx][ty + r * 8];
  }
}

// ---------------- fragment helpers ----------------

// A fragment (16x32 f16): lane group g holds K = 16g..16g+15 for row m
__device__ __forceinline__ v16h ldsA(const _Float16* At, int row, int g) {
  const _Float16* r = At + row * LDT;
  v8h lo = *(const v8h*)(r + 16 * g);
  v8h hi = *(const v8h*)(r + 16 * g + 8);
  v16h a;
  #pragma unroll
  for (int i = 0; i < 8; ++i) { a[i] = lo[i]; a[8 + i] = hi[i]; }
  return a;
}

// B fragment (32x16 f16): lane group g holds K = 16g..16g+15 for column n
__device__ __forceinline__ v16h ldsB(const _Float16* Bt, int row, int g) {
  const _Float16* r = Bt + row * LDT + 16 * g;
  v8h lo = *(const v8h*)(r);
  v8h hi = *(const v8h*)(r + 8);
  v16h b;
  #pragma unroll
  for (int i = 0; i < 8; ++i) { b[i] = lo[i]; b[8 + i] = hi[i]; }
  return b;
}

// ---------------- tile staging ----------------
// A: 64x32 halves, 256 threads -> 8 halves each
// B: 128x32 halves (from pre-transposed f16 WT, row n stride K) -> 16 halves each
// ASRC==0: A gathered from fp32 x via perm (sync, with f32->f16 convert)
// ASRC==1: A is contiguous f16 (async-capable)

template<int ASRC, int K, int N>
__device__ __forceinline__ void stage_tile(
    const void* __restrict__ Asrc, const _Float16* __restrict__ WTt,
    int k0, int nBase, int gRow, int src_b,
    int aRow, int aCol, int bRow, int bCol,
    _Float16* At, _Float16* Bt)
{
  // ---- A ----
  {
    int ka = k0 + aCol;
    _Float16* dst = &At[aRow * LDT + aCol];
    if (ASRC == 0) {
      v8h av = {};
      if (src_b >= 0 && ka < K) {
        const float* xr = (const float*)Asrc + (long)src_b * K + ka;
        v4f f0 = *(const v4f*)(xr);
        v4f f1 = *(const v4f*)(xr + 4);
        #pragma unroll
        for (int i = 0; i < 4; ++i) { av[i] = (_Float16)f0[i]; av[4 + i] = (_Float16)f1[i]; }
      }
      *(v8h*)dst = av;
    } else {
      const _Float16* ar = (const _Float16*)Asrc + (long)gRow * K + ka;   // K % 32 == 0 here
#if HAVE_ASYNC_LDS
      async_cp16(ar, dst);
#else
      *(v8h*)dst = *(const v8h*)ar;
#endif
    }
  }
  // ---- B (N-major f16; 2 x 16B per thread) ----
  {
    int gn = nBase + bRow;
    int kb = k0 + bCol;
    _Float16* dst = &Bt[bRow * LDT + bCol];
    if (gn < N) {                         // skipped rows feed only discarded C columns
      if (kb < K) {                       // K % 16 == 0 -> chunk-uniform
        const _Float16* wr = WTt + (long)gn * K + kb;
#if HAVE_ASYNC_LDS
        async_cp16(wr, dst);
        async_cp16(wr + 8, dst + 8);
#else
        *(v8h*)dst       = *(const v8h*)(wr);
        *(v8h*)(dst + 8) = *(const v8h*)(wr + 8);
#endif
      } else {
        v8h zz = {};
        *(v8h*)dst       = zz;
        *(v8h*)(dst + 8) = zz;
      }
    }
  }
}

// ---------------- grouped WMMA GEMM ----------------
// EPI: 0 = relu -> f16 store; 1 = raw f32 store; 2 = sigmoid + scatter via perm to d_out

template<int ASRC, int EPI, int K, int N>
__global__ __launch_bounds__(256) void gemm_tile(
    const void* __restrict__ Asrc, const _Float16* __restrict__ WT,
    const float* __restrict__ bias, void* __restrict__ Out,
    const int* __restrict__ perm, const int* __restrict__ offsets,
    long wStride, int biasStride)
{
  __shared__ _Float16 At[2][TILE_M * LDT];   // 2 x 5KB
  __shared__ _Float16 Bt[2][TILE_N * LDT];   // 2 x 10KB

  const int rowBase = blockIdx.x * TILE_M;
  const int total = offsets[T_];
  if (rowBase >= total) return;              // block-uniform, before any barrier
  int t = 0;
  while (offsets[t + 1] <= rowBase) ++t;     // T_=10, trivial search

  const _Float16* WTt = WT + (long)t * wStride;
  const float*    bt  = bias + (long)t * biasStride;

  const int tid   = threadIdx.x;
  const int nBase = blockIdx.y * TILE_N;

  // wave geometry: 8 waves in 2(M) x 4(N); each wave owns a 32x32 C sub-tile
  const int w    = tid >> 5;
  const int wmw  = w >> 2;                   // 0..1
  const int wnw  = w & 3;                    // 0..3
  const int lane = tid & 31;
  const int lm   = lane & 15;
  const int g    = lane >> 4;

  // staging assignment
  const int aRow = tid >> 2;                 // 0..63
  const int aCol = (tid & 3) * 8;            // 8 halves
  const int bRow = tid >> 1;                 // 0..127
  const int bCol = (tid & 1) * 16;           // 16 halves

  const int gRow = rowBase + aRow;
  int src_b = 0;
  if (ASRC == 0) src_b = perm[gRow];

  v8f acc[2][2] = {};

  constexpr int nIter = (K + TILE_K - 1) / TILE_K;

  // prologue: stage tile 0
  stage_tile<ASRC, K, N>(Asrc, WTt, 0, nBase, gRow, src_b,
                         aRow, aCol, bRow, bCol, At[0], Bt[0]);
  wait_async0();
  __syncthreads();

  int cur = 0;
  for (int it = 0; it < nIter - 1; ++it) {
    // stage next tile into the other buffer (overlaps with WMMAs below)
    stage_tile<ASRC, K, N>(Asrc, WTt, (it + 1) * TILE_K, nBase, gRow, src_b,
                           aRow, aCol, bRow, bCol, At[cur ^ 1], Bt[cur ^ 1]);

    v16h a0 = ldsA(At[cur], wmw * 32 + lm,      g);
    v16h a1 = ldsA(At[cur], wmw * 32 + 16 + lm, g);
    v16h b0 = ldsB(Bt[cur], wnw * 32 + lm,      g);
    v16h b1 = ldsB(Bt[cur], wnw * 32 + 16 + lm, g);

    acc[0][0] = __builtin_amdgcn_wmma_f32_16x16x32_f16(false, a0, false, b0, (short)0, acc[0][0], false, false);
    acc[0][1] = __builtin_amdgcn_wmma_f32_16x16x32_f16(false, a0, false, b1, (short)0, acc[0][1], false, false);
    acc[1][0] = __builtin_amdgcn_wmma_f32_16x16x32_f16(false, a1, false, b0, (short)0, acc[1][0], false, false);
    acc[1][1] = __builtin_amdgcn_wmma_f32_16x16x32_f16(false, a1, false, b1, (short)0, acc[1][1], false, false);

    wait_async0();     // async stage of next buffer must land before the barrier
    __syncthreads();   // next-buffer visible; current-buffer reads retired
    cur ^= 1;
  }

  // last tile: compute only, no trailing barrier needed
  {
    v16h a0 = ldsA(At[cur], wmw * 32 + lm,      g);
    v16h a1 = ldsA(At[cur], wmw * 32 + 16 + lm, g);
    v16h b0 = ldsB(Bt[cur], wnw * 32 + lm,      g);
    v16h b1 = ldsB(Bt[cur], wnw * 32 + 16 + lm, g);

    acc[0][0] = __builtin_amdgcn_wmma_f32_16x16x32_f16(false, a0, false, b0, (short)0, acc[0][0], false, false);
    acc[0][1] = __builtin_amdgcn_wmma_f32_16x16x32_f16(false, a0, false, b1, (short)0, acc[0][1], false, false);
    acc[1][0] = __builtin_amdgcn_wmma_f32_16x16x32_f16(false, a1, false, b0, (short)0, acc[1][0], false, false);
    acc[1][1] = __builtin_amdgcn_wmma_f32_16x16x32_f16(false, a1, false, b1, (short)0, acc[1][1], false, false);
  }

  // ---- epilogue ----
  #pragma unroll
  for (int mi = 0; mi < 2; ++mi) {
    #pragma unroll
    for (int nj = 0; nj < 2; ++nj) {
      int n = nBase + wnw * 32 + nj * 16 + lm;
      if (n < N) {
        float bv = bt[n];
        #pragma unroll
        for (int i = 0; i < 8; ++i) {
          int m = wmw * 32 + mi * 16 + 8 * g + i;  // C layout: VGPR i -> M=i / M=i+8
          int p = rowBase + m;
          float v = acc[mi][nj][i] + bv;
          if (EPI == 0) {
            v = v > 0.f ? v : 0.f;
            ((_Float16*)Out)[(long)p * N + n] = (_Float16)v;
          } else if (EPI == 1) {
            ((float*)Out)[(long)p * N + n] = v;
          } else {
            int ob = perm[p];
            if (ob >= 0) {
              float s = 1.f / (1.f + __expf(-v));
              ((float*)Out)[(long)ob * N + n] = s;
            }
          }
        }
      }
    }
  }
}

// ---------------- reparameterization + output of mu / log_sigma ----------------

__global__ void k_z(const float* __restrict__ enc, const float* __restrict__ eps,
                    const int* __restrict__ perm, _Float16* __restrict__ z,
                    float* __restrict__ outMu, float* __restrict__ outLs)
{
  int p = blockIdx.x;       // padded-permuted row
  int i = threadIdx.x;      // 0..127
  int b = perm[p];
  if (b < 0) { z[(long)p * L_ + i] = (_Float16)0.f; return; }  // block-uniform
  const float* e = enc + (long)p * (2 * L_);
  float mu = e[i];
  float ls = e[L_ + i];
  outMu[(long)b * L_ + i] = mu;
  outLs[(long)b * L_ + i] = ls;
  z[(long)p * L_ + i] = (_Float16)(mu + __expf(ls) * eps[(long)b * L_ + i]);
}

// ---------------- host launcher ----------------

extern "C" void kernel_launch(void* const* d_in, const int* in_sizes, int n_in,
                              void* d_out, int out_size, void* d_ws, size_t ws_size,
                              hipStream_t stream)
{
  const float* x      = (const float*)d_in[0];
  const int*   task   = (const int*)  d_in[1];
  const float* eps    = (const float*)d_in[2];
  const float* enc_W1 = (const float*)d_in[3];
  const float* enc_b1 = (const float*)d_in[4];
  const float* enc_W2 = (const float*)d_in[5];
  const float* enc_b2 = (const float*)d_in[6];
  const float* enc_W3 = (const float*)d_in[7];
  const float* enc_b3 = (const float*)d_in[8];
  const float* enc_W4 = (const float*)d_in[9];
  const float* enc_b4 = (const float*)d_in[10];
  const float* dec_Ws1= (const float*)d_in[11];
  const float* dec_bs1= (const float*)d_in[12];
  const float* dec_Ws2= (const float*)d_in[13];
  const float* dec_bs2= (const float*)d_in[14];
  const float* dec_Wh1= (const float*)d_in[15];
  const float* dec_bh1= (const float*)d_in[16];
  const float* dec_Wh2= (const float*)d_in[17];
  const float* dec_bh2= (const float*)d_in[18];

  // workspace carve-up
  char* ws = (char*)d_ws;
  int* counts  = (int*)(ws + 0);
  int* cursor  = (int*)(ws + 64);
  int* offsets = (int*)(ws + 128);
  int* perm    = (int*)(ws + 256);
  size_t off = 256 + sizeof(int) * (size_t)BPAD;
  off = (off + 255) & ~(size_t)255;
  _Float16* actA = (_Float16*)(ws + off); off += (size_t)BPAD * H_ * 2;
  _Float16* actB = (_Float16*)(ws + off); off += (size_t)BPAD * H_ * 2;
  float*    encb = (float*)   (ws + off); off += (size_t)BPAD * 2 * L_ * 4;
  _Float16* zbuf = (_Float16*)(ws + off); off += (size_t)BPAD * L_ * 2;
  // transposed f16 weights (N-major)
  _Float16* wt1  = (_Float16*)(ws + off); off += (size_t)T_ * IN_ * H_ * 2;
  _Float16* wt2  = (_Float16*)(ws + off); off += (size_t)T_ * H_ * H_ * 2;
  _Float16* wt3  = (_Float16*)(ws + off); off += (size_t)T_ * H_ * H_ * 2;
  _Float16* wt4  = (_Float16*)(ws + off); off += (size_t)T_ * H_ * 2 * L_ * 2;
  _Float16* wts1 = (_Float16*)(ws + off); off += (size_t)L_ * H_ * 2;
  _Float16* wts2 = (_Float16*)(ws + off); off += (size_t)H_ * H_ * 2;
  _Float16* wth1 = (_Float16*)(ws + off); off += (size_t)T_ * H_ * H_ * 2;
  _Float16* wth2 = (_Float16*)(ws + off); off += (size_t)T_ * H_ * IN_ * 2;

  float* out_x  = (float*)d_out;                      // (B, IN)
  float* out_mu = out_x  + (size_t)B_ * IN_;          // (B, L)
  float* out_ls = out_mu + (size_t)B_ * L_;           // (B, L)

  // ---- pre-transpose all weights to f16 N-major ----
  dim3 tb(32, 8);
  k_transpose<<<dim3(32, 25, T_), tb, 0, stream>>>(enc_W1, wt1, IN_, H_);     // K=784,  N=1024
  k_transpose<<<dim3(32, 32, T_), tb, 0, stream>>>(enc_W2, wt2, H_, H_);
  k_transpose<<<dim3(32, 32, T_), tb, 0, stream>>>(enc_W3, wt3, H_, H_);
  k_transpose<<<dim3(8,  32, T_), tb, 0, stream>>>(enc_W4, wt4, H_, 2 * L_);  // K=1024, N=256
  k_transpose<<<dim3(32, 4,  1 ), tb, 0, stream>>>(dec_Ws1, wts1, L_, H_);    // K=128,  N=1024
  k_transpose<<<dim3(32, 32, 1 ), tb, 0, stream>>>(dec_Ws2, wts2, H_, H_);
  k_transpose<<<dim3(32, 32, T_), tb, 0, stream>>>(dec_Wh1, wth1, H_, H_);
  k_transpose<<<dim3(25, 32, T_), tb, 0, stream>>>(dec_Wh2, wth2, H_, IN_);   // K=1024, N=784

  // ---- build task-sorted, tile-padded permutation ----
  k_init   <<<1,      256, 0, stream>>>(counts, cursor, perm);
  k_count  <<<B_/256, 256, 0, stream>>>(task, counts);
  k_scan   <<<1,      1,   0, stream>>>(counts, offsets);
  k_scatter<<<B_/256, 256, 0, stream>>>(task, offsets, cursor, perm);

  dim3 blk(256);
  dim3 gH(MAX_ROWTILES, H_ / TILE_N);                    // 138 x 8
  dim3 gE(MAX_ROWTILES, (2 * L_ + TILE_N - 1) / TILE_N); // 138 x 2
  dim3 gO(MAX_ROWTILES, (IN_ + TILE_N - 1) / TILE_N);    // 138 x 7

  // encoder (per-task weights)
  gemm_tile<0,0,IN_,H_>   <<<gH, blk, 0, stream>>>(x,    wt1, enc_b1, actA, perm, offsets, (long)IN_ * H_,    H_);
  gemm_tile<1,0,H_,H_>    <<<gH, blk, 0, stream>>>(actA, wt2, enc_b2, actB, perm, offsets, (long)H_ * H_,     H_);
  gemm_tile<1,0,H_,H_>    <<<gH, blk, 0, stream>>>(actB, wt3, enc_b3, actA, perm, offsets, (long)H_ * H_,     H_);
  gemm_tile<1,1,H_,2*L_>  <<<gE, blk, 0, stream>>>(actA, wt4, enc_b4, encb, perm, offsets, (long)H_ * 2 * L_, 2*L_);

  // reparameterize + emit mu / log_sigma
  k_z<<<BPAD, L_, 0, stream>>>(encb, eps, perm, zbuf, out_mu, out_ls);

  // decoder (shared then per-task weights)
  gemm_tile<1,0,L_,H_>    <<<gH, blk, 0, stream>>>(zbuf, wts1, dec_bs1, actA, perm, offsets, 0L,              0);
  gemm_tile<1,0,H_,H_>    <<<gH, blk, 0, stream>>>(actA, wts2, dec_bs2, actB, perm, offsets, 0L,              0);
  gemm_tile<1,0,H_,H_>    <<<gH, blk, 0, stream>>>(actB, wth1, dec_bh1, actA, perm, offsets, (long)H_ * H_,   H_);
  gemm_tile<1,2,H_,IN_>   <<<gO, blk, 0, stream>>>(actA, wth2, dec_bh2, out_x, perm, offsets, (long)H_ * IN_, IN_);
}
